// MultilayerGRU_47794396069902
// MI455X (gfx1250) — matrix operand order repaired and checked by reading, not compiled, not run on hardware
//
#include <hip/hip_runtime.h>
#include <hip/hip_bf16.h>
#include <math.h>

// ---------------- CDNA5 vector types ----------------
typedef __attribute__((ext_vector_type(16))) __bf16       v16bf;
typedef __attribute__((ext_vector_type(8)))  float        v8f;
typedef __attribute__((ext_vector_type(8)))  unsigned int v8u;
typedef __attribute__((ext_vector_type(4)))  int          v4i;

#define AS1 __attribute__((address_space(1)))
#define AS3 __attribute__((address_space(3)))

// ---------------- problem constants ----------------
constexpr int kB = 128, kS = 256, kI = 256, kH = 1024, kO = 256;
constexpr int SB = kS * kB;   // 32768 flattened (s,b) rows

// ---------------- GEMM tile parameters ----------------
constexpr int BM = 32;        // block M tile
constexpr int BN = 128;       // block N tile (8 waves: 2M x 4N, wave = 16x32)
constexpr int BK = 32;        // K chunk (one wmma_f32_16x16x32_bf16 depth)
constexpr int APITCH = 36;    // A LDS pitch (halves), padded to dodge bank conflicts
constexpr int BSLAB  = BN * BK;  // 4096 halves = 8KB contiguous B slab

// ---------------- helpers ----------------
__device__ __forceinline__ unsigned short f32_to_bf16(float f) {
  unsigned u = __builtin_bit_cast(unsigned, f);
  unsigned r = 0x7FFFu + ((u >> 16) & 1u);   // round-to-nearest-even
  return (unsigned short)((u + r) >> 16);
}
__device__ __forceinline__ unsigned pack2_bf16(float lo, float hi) {
#if __has_builtin(__builtin_amdgcn_cvt_pk_bf16_f32)
  auto p = __builtin_amdgcn_cvt_pk_bf16_f32(lo, hi);   // single v_cvt_pk_bf16_f32
  return __builtin_bit_cast(unsigned, p);
#else
  return (unsigned)f32_to_bf16(lo) | ((unsigned)f32_to_bf16(hi) << 16);
#endif
}
__device__ __forceinline__ float sigmoid_(float x) {
  return 1.0f / (1.0f + __expf(-x));
}

// A fragment: 16x32 bf16, documented CDNA5 layout.
__device__ __forceinline__ v16bf load_a_frag(const unsigned short* As, int row0) {
  int lane = threadIdx.x & 31;
  int m    = lane & 15;
  int klo  = (lane < 16) ? 0 : 8;
  const unsigned short* p = As + (row0 + m) * APITCH;
  v8u u;
#pragma unroll
  for (int v = 0; v < 4; ++v) u[v]     = *(const unsigned*)(p + klo + 2 * v);
#pragma unroll
  for (int v = 0; v < 4; ++v) u[4 + v] = *(const unsigned*)(p + 16 + klo + 2 * v);
  return __builtin_bit_cast(v16bf, u);
}

// B fragment: 32x16 bf16; LDS slab is [n][k] with pitch 32 halves (contiguous async copy).
__device__ __forceinline__ v16bf load_b_frag(const unsigned short* Bs, int n0) {
  int lane = threadIdx.x & 31;
  int n    = n0 + (lane & 15);
  int kh   = (lane < 16) ? 0 : 8;
  const unsigned short* p = Bs + n * BK + kh;
  v8u u;
#pragma unroll
  for (int j = 0; j < 8; ++j) u[j] = *(const unsigned*)(p + ((j < 4) ? 2 * j : 16 + 2 * (j - 4)));
  return __builtin_bit_cast(v16bf, u);
}

// Async-DMA one contiguous 8KB B slab (blocked weight layout) into LDS.
// slab index = kc * nblocks + nblk ; tracked by ASYNCcnt.
__device__ __forceinline__ void stage_b_async(const unsigned short* __restrict__ WB,
                                              unsigned short* Bs, int kc, int nblk,
                                              int nblocks) {
  const unsigned short* src = WB + ((size_t)kc * nblocks + nblk) * BSLAB;
  int tid = threadIdx.x;
#pragma unroll
  for (int it = 0; it < 2; ++it) {
    int off = (it * 256 + tid) * 8;   // halves; 8 halves = 16B per lane
    __builtin_amdgcn_global_load_async_to_lds_b128(
        (AS1 v4i*)(src + off), (AS3 v4i*)(Bs + off), 0, 0);
  }
}

// A-row source resolver: MODE 0 = x in (B,S,I) layout; MODE 1/3 = row-major.
template<int MODE>
__device__ __forceinline__ const float* a_row(const float* A, int m, int K) {
  if constexpr (MODE == 0) {
    int b = m % kB, s = m / kB;
    return A + ((size_t)b * kS + s) * K;
  } else {
    return A + (size_t)m * K;
  }
}

// ---------------- non-recurrent WMMA GEMMs ----------------
// MODE 0: xg = X @ WxgT  (A = x in (B,S,I) layout)   Out = acc
// MODE 1: xg = hs0 @ WxgT (row-major)                Out = acc
// MODE 3: head: Out = sigmoid(acc + bhy) scattered to (B,S,O)
template<int MODE>
__global__ __launch_bounds__(256) void gemm_proj_kernel(
    const float* __restrict__ A,
    const unsigned short* __restrict__ WB,   // blocked bf16 weights
    float* __restrict__ Out,
    const float* __restrict__ addv,          // MODE3: bhy
    int N, int K)
{
  __shared__ unsigned short As[BM * APITCH];
  __shared__ unsigned short Bs[BSLAB];

  const int tid  = threadIdx.x;
  const int m0   = blockIdx.x * BM;
  const int nblk = blockIdx.y;
  const int n0   = nblk * BN;
  const int nblocks = N / BN;
  const int wave = tid >> 5, lane = tid & 31;
  const int wm   = wave >> 2;   // 0..1
  const int wn   = wave & 3;    // 0..3

  v8f acc[2];
#pragma unroll
  for (int t = 0; t < 2; ++t) acc[t] = v8f{0.f,0.f,0.f,0.f,0.f,0.f,0.f,0.f};

  const int nk = K / BK;
  for (int kc = 0; kc < nk; ++kc) {
    const int k0 = kc * BK;
    stage_b_async(WB, Bs, kc, nblk, nblocks);   // DMA weights while VALU converts A
#pragma unroll
    for (int it = 0; it < 2; ++it) {
      int idx = it * 256 + tid;       // (row, k-pair)
      int row = idx >> 4;
      int kp  = idx & 15;
      const float* src = a_row<MODE>(A, m0 + row, K);
      *(unsigned*)&As[row * APITCH + 2 * kp] =
          pack2_bf16(src[k0 + 2 * kp], src[k0 + 2 * kp + 1]);
    }
    __builtin_amdgcn_s_wait_asynccnt(0);
    __syncthreads();

    v16bf afrag = load_a_frag(As, wm * 16);
#pragma unroll
    for (int t = 0; t < 2; ++t) {
      v16bf bfrag = load_b_frag(Bs, wn * 32 + t * 16);
      acc[t] = __builtin_amdgcn_wmma_f32_16x16x32_bf16(
          false, afrag, false, bfrag, (short)0, acc[t], false, false);
    }
    __syncthreads();
  }

  const int mrow0 = m0 + wm * 16 + ((lane < 16) ? 0 : 8);
#pragma unroll
  for (int t = 0; t < 2; ++t) {
    int nc = n0 + wn * 32 + t * 16 + (lane & 15);
#pragma unroll
    for (int v = 0; v < 8; ++v) {
      int   m  = mrow0 + v;
      float av = acc[t][v];
      if constexpr (MODE == 3) {
        int s = m / kB, b = m % kB;
        Out[((size_t)b * kS + s) * kO + nc] = sigmoid_(av + addv[nc]);
      } else {
        Out[(size_t)m * N + nc] = av;
      }
    }
  }
}

// ---------------- persistent GRU layer kernel ----------------
// One launch runs all kS timesteps; 32 co-resident workgroups sync through a
// monotonically increasing global arrival counter between steps.
__device__ __forceinline__ void grid_barrier(unsigned* bar, unsigned target) {
  __threadfence();                 // make this block's hs writes device-visible
  __syncthreads();
  if (threadIdx.x == 0) {
    __hip_atomic_fetch_add(bar, 1u, __ATOMIC_ACQ_REL, __HIP_MEMORY_SCOPE_AGENT);
    while (__hip_atomic_load(bar, __ATOMIC_ACQUIRE, __HIP_MEMORY_SCOPE_AGENT) < target)
      __builtin_amdgcn_s_sleep(1);
  }
  __syncthreads();
}

__global__ __launch_bounds__(256) void gru_layer_kernel(
    const float* __restrict__ h0l,           // layer slice of h0 (row stride 2H)
    float* __restrict__ hs,                  // (S,B,H) hidden sequence (out)
    const float* __restrict__ xg,            // (S,B,H) input projection
    const unsigned short* __restrict__ WB,   // blocked bf16 WhgT
    const float* __restrict__ bhg,
    const float* __restrict__ Whz,
    const float* __restrict__ Whr,
    const float* __restrict__ bhz,
    const float* __restrict__ xzv,
    const float* __restrict__ xrv,
    unsigned* __restrict__ bar)
{
  __shared__ unsigned short As[BM * APITCH];
  __shared__ unsigned short Bs[BSLAB];
  __shared__ float zsh[BM];
  __shared__ float rsh[BM];

  const int tid  = threadIdx.x;
  const int m0   = blockIdx.x * BM;          // batch-row tile
  const int nblk = blockIdx.y;
  const int n0   = nblk * BN;
  const int nblocks = kH / BN;               // 8
  const int wave = tid >> 5, lane = tid & 31;
  const int wm   = wave >> 2, wn = wave & 3;
  const unsigned nwg = gridDim.x * gridDim.y;  // 32

  for (int s = 0; s < kS; ++s) {
    const float* hp = (s == 0) ? h0l : (hs + (size_t)(s - 1) * kB * kH);
    const int hstr  = (s == 0) ? 2 * kH : kH;
    const int sbase = s * kB;

    // ---- scalar gates z,r for this block's 32 batch rows ----
#pragma unroll
    for (int i = 0; i < 4; ++i) {
      int row = wave * 4 + i;
      int m   = m0 + row;
      const float* h = hp + (size_t)m * hstr;
      float s1 = 0.f, s2 = 0.f;
      for (int k = lane; k < kH; k += 32) {
        float hv = h[k];
        s1 += hv * Whz[k];
        s2 += hv * Whr[k];
      }
#pragma unroll
      for (int off = 16; off > 0; off >>= 1) {
        s1 += __shfl_xor(s1, off, 32);
        s2 += __shfl_xor(s2, off, 32);
      }
      if (lane == 0) {
        zsh[row] = sigmoid_(xzv[sbase + m] + s1 + bhz[0]);
        rsh[row] = sigmoid_(xrv[sbase + m] + s2);
      }
    }
    __syncthreads();

    // ---- (r*h) @ WhgT via WMMA ----
    v8f acc[2];
#pragma unroll
    for (int t = 0; t < 2; ++t) acc[t] = v8f{0.f,0.f,0.f,0.f,0.f,0.f,0.f,0.f};

    for (int kc = 0; kc < kH / BK; ++kc) {
      const int k0 = kc * BK;
      stage_b_async(WB, Bs, kc, nblk, nblocks);
#pragma unroll
      for (int it = 0; it < 2; ++it) {
        int idx = it * 256 + tid;
        int row = idx >> 4;
        int kp  = idx & 15;
        const float* src = hp + (size_t)(m0 + row) * hstr;
        float rr = rsh[row];
        *(unsigned*)&As[row * APITCH + 2 * kp] =
            pack2_bf16(src[k0 + 2 * kp] * rr, src[k0 + 2 * kp + 1] * rr);
      }
      __builtin_amdgcn_s_wait_asynccnt(0);
      __syncthreads();

      v16bf afrag = load_a_frag(As, wm * 16);
#pragma unroll
      for (int t = 0; t < 2; ++t) {
        v16bf bfrag = load_b_frag(Bs, wn * 32 + t * 16);
        acc[t] = __builtin_amdgcn_wmma_f32_16x16x32_bf16(
            false, afrag, false, bfrag, (short)0, acc[t], false, false);
      }
      __syncthreads();
    }

    // ---- epilogue: h_new = z*h + (1-z)*tanh(acc + xg_t + bhg) ----
    const int mrow0 = m0 + wm * 16 + ((lane < 16) ? 0 : 8);
#pragma unroll
    for (int t = 0; t < 2; ++t) {
      int nc = n0 + wn * 32 + t * 16 + (lane & 15);
#pragma unroll
      for (int v = 0; v < 8; ++v) {
        int   m  = mrow0 + v;
        float hv = hp[(size_t)m * hstr + nc];
        float z  = zsh[m - m0];
        float g  = tanhf(acc[t][v] + xg[(size_t)(sbase + m) * kH + nc] + bhg[nc]);
        hs[(size_t)(sbase + m) * kH + nc] = z * hv + (1.f - z) * g;
      }
    }

    grid_barrier(bar, nwg * (unsigned)(s + 1));
  }
}

// ---------------- scalar input projections xz, xr (time-invariant) ----------------
template<bool XLAYOUT>
__global__ __launch_bounds__(256) void scalar_proj_kernel(
    const float* __restrict__ A, const float* __restrict__ Wz,
    const float* __restrict__ Wr, const float* __restrict__ bxr,
    float* __restrict__ xz, float* __restrict__ xr, int K)
{
  int m    = (blockIdx.x * blockDim.x + threadIdx.x) >> 5;
  int lane = threadIdx.x & 31;
  const float* row;
  if constexpr (XLAYOUT) {
    int b = m % kB, s = m / kB;
    row = A + ((size_t)b * kS + s) * K;
  } else {
    row = A + (size_t)m * K;
  }
  float s1 = 0.f, s2 = 0.f;
  for (int k = lane; k < K; k += 32) {
    float v = row[k];
    s1 += v * Wz[k];
    s2 += v * Wr[k];
  }
#pragma unroll
  for (int off = 16; off > 0; off >>= 1) {
    s1 += __shfl_xor(s1, off, 32);
    s2 += __shfl_xor(s2, off, 32);
  }
  if (lane == 0) { xz[m] = s1; xr[m] = s2 + bxr[0]; }
}

// ---------------- weight convert: blocked bf16, DMA-friendly ----------------
// WT slab layout: [k/32][n/128][n%128][k%32]  (each slab 8KB contiguous)
__global__ __launch_bounds__(256) void conv_blocked_kernel(
    const float* __restrict__ W, unsigned short* __restrict__ WT, int N, int K)
{
  int idx = blockIdx.x * 256 + threadIdx.x;   // [0, N*K)
  int k = idx / N, n = idx % N;
  size_t slab = (size_t)(k >> 5) * (N >> 7) + (n >> 7);
  WT[slab * BSLAB + (n & 127) * BK + (k & 31)] = f32_to_bf16(W[(size_t)n * K + k]);
}

// ---------------- final hidden state (B, L=2, H) ----------------
__global__ __launch_bounds__(256) void copy_hidden_kernel(
    const float* __restrict__ hs0, const float* __restrict__ hs1, float* __restrict__ dst)
{
  int idx = blockIdx.x * 256 + threadIdx.x;   // [0, 2*B*H)
  int hh = idx % kH;
  int l  = (idx / kH) & 1;
  int b  = idx / (2 * kH);
  const float* src = l ? hs1 : hs0;
  dst[idx] = src[((size_t)(kS - 1) * kB + b) * kH + hh];
}

// ---------------- launcher ----------------
extern "C" void kernel_launch(void* const* d_in, const int* in_sizes, int n_in,
                              void* d_out, int out_size, void* d_ws, size_t ws_size,
                              hipStream_t stream) {
  const float* x    = (const float*)d_in[0];
  const float* h0   = (const float*)d_in[1];
  const float* Wxz0 = (const float*)d_in[2];
  const float* Whz0 = (const float*)d_in[3];
  const float* bhz0 = (const float*)d_in[4];
  const float* Wxr0 = (const float*)d_in[5];
  const float* bxr0 = (const float*)d_in[6];
  const float* Whr0 = (const float*)d_in[7];
  const float* Wxg0 = (const float*)d_in[8];
  const float* Whg0 = (const float*)d_in[9];
  const float* bhg0 = (const float*)d_in[10];
  const float* Wxz1 = (const float*)d_in[11];
  const float* Whz1 = (const float*)d_in[12];
  const float* bhz1 = (const float*)d_in[13];
  const float* Wxr1 = (const float*)d_in[14];
  const float* bxr1 = (const float*)d_in[15];
  const float* Whr1 = (const float*)d_in[16];
  const float* Wxg1 = (const float*)d_in[17];
  const float* Whg1 = (const float*)d_in[18];
  const float* bhg1 = (const float*)d_in[19];
  const float* Why  = (const float*)d_in[20];
  const float* bhy  = (const float*)d_in[21];
  float* out = (float*)d_out;

  size_t off = 0;
  auto carve = [&](size_t bytes) -> void* {
    void* p = (char*)d_ws + off;
    off += (bytes + 255) & ~(size_t)255;
    return p;
  };
  float* xg  = (float*)carve((size_t)SB * kH * 4);
  float* hs0 = (float*)carve((size_t)SB * kH * 4);
  float* hs1 = (float*)carve((size_t)SB * kH * 4);
  float* xz  = (float*)carve((size_t)SB * 4);
  float* xr  = (float*)carve((size_t)SB * 4);
  unsigned* bar = (unsigned*)carve(256);
  unsigned short* WhgT0 = (unsigned short*)carve((size_t)kH * kH * 2);
  unsigned short* WhgT1 = (unsigned short*)carve((size_t)kH * kH * 2);
  unsigned short* WxgT0 = (unsigned short*)carve((size_t)kI * kH * 2);
  unsigned short* WxgT1 = (unsigned short*)carve((size_t)kH * kH * 2);
  unsigned short* WhyT  = (unsigned short*)carve((size_t)kH * kO * 2);

  (void)hipMemsetAsync(bar, 0, 256, stream);   // reset grid-barrier counters each call

  // ---- one-time: weights -> blocked bf16 ----
  conv_blocked_kernel<<<(kH * kH) / 256, 256, 0, stream>>>(Whg0, WhgT0, kH, kH);
  conv_blocked_kernel<<<(kH * kH) / 256, 256, 0, stream>>>(Whg1, WhgT1, kH, kH);
  conv_blocked_kernel<<<(kH * kI) / 256, 256, 0, stream>>>(Wxg0, WxgT0, kH, kI);
  conv_blocked_kernel<<<(kH * kH) / 256, 256, 0, stream>>>(Wxg1, WxgT1, kH, kH);
  conv_blocked_kernel<<<(kO * kH) / 256, 256, 0, stream>>>(Why,  WhyT,  kO, kH);

  // ---- layer 0 ----
  scalar_proj_kernel<true><<<SB / 8, 256, 0, stream>>>(x, Wxz0, Wxr0, bxr0, xz, xr, kI);
  gemm_proj_kernel<0><<<dim3(SB / BM, kH / BN), 256, 0, stream>>>(
      x, WxgT0, xg, nullptr, kH, kI);
  gru_layer_kernel<<<dim3(kB / BM, kH / BN), 256, 0, stream>>>(
      h0, hs0, xg, WhgT0, bhg0, Whz0, Whr0, bhz0, xz, xr, bar + 0);

  // ---- layer 1 ----
  scalar_proj_kernel<false><<<SB / 8, 256, 0, stream>>>(hs0, Wxz1, Wxr1, bxr1, xz, xr, kH);
  gemm_proj_kernel<1><<<dim3(SB / BM, kH / BN), 256, 0, stream>>>(
      hs0, WxgT1, xg, nullptr, kH, kH);
  gru_layer_kernel<<<dim3(kB / BM, kH / BN), 256, 0, stream>>>(
      h0 + kH, hs1, xg, WhgT1, bhg1, Whz1, Whr1, bhz1, xz, xr, bar + 1);

  // ---- output head ----
  gemm_proj_kernel<3><<<dim3(SB / BM, kO / BN), 256, 0, stream>>>(
      hs1, WhyT, out, bhy, kO, kH);

  // ---- final hidden states ----
  copy_hidden_kernel<<<(2 * kB * kH) / 256, 256, 0, stream>>>(
      hs0, hs1, out + (size_t)kB * kS * kO);
}
